// SplitLayerNeRF_15788299780562
// MI455X (gfx1250) — matrix operand order repaired and checked by reading, not compiled, not run on hardware
//
#include <hip/hip_runtime.h>
#include <cstddef>
#include <cstdint>

#define N_PTS    131072
#define IN_CH    63
#define IN_VIEWS 27
#define WIDTH    256
#define NHEADS   4

typedef __attribute__((ext_vector_type(16))) _Float16 v16h;
typedef __attribute__((ext_vector_type(8)))  _Float16 v8h;
typedef __attribute__((ext_vector_type(8)))  float    v8f;

// LDS: actA(128x320 f16) | actB(128x320 f16) | wtb0(32768 h) | wtb1(32768 h) = 288KB
#define ACT_HALVES   40960            // 128*320
#define WT_HALVES    32768
#define SMEM_HALVES  (2 * ACT_HALVES + 2 * WT_HALVES)
#define SMEM_BYTES   (SMEM_HALVES * 2)

// ---- panel layout in workspace (units: halves) ----
#define P_W0        0u                              /* 256x64   = 16384 */
#define P_WS(l,h2)  (16384u + ((l)*2u+(h2))*32768u) /* 6 x 128x256      */
#define P_HEADBASE  212992u
#define PH_SIZE     391168u
#define PH_TW0(q)   ((q)*20480u)                    /* 4 x 64x320       */
#define PH_TWS(l,h2) (81920u + ((l)*2u+(h2))*32768u)/* 6 x 128x256      */
#define PH_ALPHA    278528u                         /* 16x256 = 4096    */
#define PH_FEAT(h2) (282624u + (h2)*32768u)         /* 2 x 128x256      */
#define PH_VIEWS(h2)(348160u + (h2)*20480u)         /* 2 x 64x320       */
#define PH_RGB      389120u                         /* 16x128 = 2048    */
#define PANELS_TOTAL (P_HEADBASE + 4u*PH_SIZE)      /* 1777664 halves   */

__host__ __device__ __forceinline__ int swz(int row, int col) {
  return ((((col >> 3) ^ (row & 7)) << 3) | (col & 7));
}

// A fragment (16x32 f16): lane<16 -> M=lane, K=k0+0..15; lane>=16 -> K=k0+16..31
__device__ __forceinline__ v16h load_a_frag(const _Float16* act, int stride,
                                            int m, int k0, int khalf) {
  const _Float16* p0 = act + m * stride + swz(m, k0 + 2 * khalf);
  const _Float16* p1 = act + m * stride + swz(m, k0 + 2 * khalf + 8);
  v8h lo = *(const v8h*)p0;
  v8h hi = *(const v8h*)p1;
  v16h r;
#pragma unroll
  for (int e = 0; e < 8; ++e) { r[e] = lo[e]; r[e + 8] = hi[e]; }
  return r;
}

// B fragment (32x16 f16): lane<16 -> N=lane, K=k0+0..15; lane>=16 -> K=k0+16..31
__device__ __forceinline__ v16h load_b_frag(const _Float16* wt, int kpad,
                                            int n, int k0, int klo) {
  const _Float16* p0 = wt + n * kpad + swz(n, k0 + klo);
  const _Float16* p1 = wt + n * kpad + swz(n, k0 + klo + 8);
  v8h lo = *(const v8h*)p0;
  v8h hi = *(const v8h*)p1;
  v16h r;
#pragma unroll
  for (int e = 0; e < 8; ++e) { r[e] = lo[e]; r[e + 8] = hi[e]; }
  return r;
}

// Async global->LDS flat copy (gfx1250 ASYNCcnt path); all 256 threads, 16B each.
__device__ __forceinline__ void stage_async(const _Float16* __restrict__ g,
                                            _Float16* l, int halves) {
  uint32_t ldsoff = (uint32_t)(uintptr_t)l + (uint32_t)threadIdx.x * 16u;
  const char* gp = (const char*)g + (size_t)threadIdx.x * 16u;
  const int bytes = halves * 2;
  for (int off = threadIdx.x * 16; off < bytes; off += blockDim.x * 16) {
    asm volatile("global_load_async_to_lds_b128 %0, %1, off"
                 :: "v"(ldsoff), "v"(gp) : "memory");
    ldsoff += blockDim.x * 16u;
    gp += (size_t)blockDim.x * 16u;
  }
}

__device__ __forceinline__ void wait_async_and_sync() {
  asm volatile("s_wait_asynccnt 0x0" ::: "memory");
  __syncthreads();
}

// One dense layer stage, fully unrolled. A cached in registers (KT frags), bias
// preloaded (NT regs), B batched per tile (KT frags loaded, then KT WMMAs) so
// ds_loads pipeline ahead of the WMMA chain with partial dscnt waits.
template <int KT, int NT, bool RELU>
__device__ void mlp_layer(const _Float16* A, int sA, _Float16* B, int sB,
                          const _Float16* wt, int kpad,
                          const float* __restrict__ bias, int nt0) {
  const int lane  = threadIdx.x & 31;
  const int wave  = threadIdx.x >> 5;
  const int m0    = wave * 16;
  const int hi16  = lane >> 4;
  const int mlane = lane & 15;
  const int khalf = hi16 * 8;
  const int klo   = hi16 * 16;
  const int nlane = lane & 15;
  const int mbase = hi16 * 8;

  float bv[NT];
#pragma unroll
  for (int t = 0; t < NT; ++t) bv[t] = bias[(nt0 + t) * 16 + nlane];

  v16h a[KT];
#pragma unroll
  for (int kt = 0; kt < KT; ++kt)
    a[kt] = load_a_frag(A, sA, m0 + mlane, kt * 32, khalf);

#pragma unroll
  for (int t = 0; t < NT; ++t) {
    v16h bf[KT];
#pragma unroll
    for (int kt = 0; kt < KT; ++kt)
      bf[kt] = load_b_frag(wt, kpad, t * 16 + nlane, kt * 32, klo);
    v8f acc = {};
#pragma unroll
    for (int kt = 0; kt < KT; ++kt)
      acc = __builtin_amdgcn_wmma_f32_16x16x32_f16(false, a[kt], false, bf[kt], (short)0,
                                                   acc, false, false);
    const int col = (nt0 + t) * 16 + nlane;
#pragma unroll
    for (int r = 0; r < 8; ++r) {
      int m = m0 + mbase + r;
      float v = acc[r] + bv[t];
      if (RELU) v = fmaxf(v, 0.f);
      B[m * sB + swz(m, col)] = (_Float16)v;
    }
  }
}

// ---------------- weight panel prep ----------------
struct PanelDesc {
  const float* src;
  unsigned dstoff;   // halves
  short Kin, Ntot, n0, nthis, Npad, kpad;
};
struct PanelTable { PanelDesc d[72]; int count; };

__global__ void prep_panels_kernel(PanelTable tab, _Float16* __restrict__ panels) {
  if ((int)blockIdx.x >= tab.count) return;
  const PanelDesc pd = tab.d[blockIdx.x];
  _Float16* dst = panels + pd.dstoff;
  const int total = (int)pd.Npad * pd.kpad;
  for (int i = threadIdx.x; i < total; i += blockDim.x) dst[i] = (_Float16)0.f;
  __syncthreads();
  const int fill = (int)pd.Kin * pd.nthis;
  for (int i = threadIdx.x; i < fill; i += blockDim.x) {
    int k = i / pd.nthis;
    int n = i - k * pd.nthis;
    dst[n * pd.kpad + swz(n, k)] = (_Float16)pd.src[(size_t)k * pd.Ntot + pd.n0 + n];
  }
}

// ---------------- routing ----------------
__global__ void route_init_kernel(int* counts) {
  if (threadIdx.x < NHEADS) counts[threadIdx.x] = 0;
}

__global__ void route_scatter_kernel(const int* __restrict__ head_idx,
                                     int* __restrict__ counts,
                                     int* __restrict__ lists) {
  int i = blockIdx.x * blockDim.x + threadIdx.x;
  if (i < N_PTS) {
    int h = head_idx[i] & 3;
    int pos = atomicAdd(&counts[h], 1);
    lists[(size_t)h * N_PTS + pos] = i;
  }
}

// ---------------- trunk ----------------
__global__ __launch_bounds__(256) void trunk_kernel(
    const float* __restrict__ x, const float* __restrict__ b0,
    const float* __restrict__ bs, const _Float16* __restrict__ panels,
    _Float16* __restrict__ h_out) {
  extern __shared__ _Float16 smem[];
  _Float16* actA = smem;
  _Float16* actB = smem + ACT_HALVES;
  _Float16* wtb0 = smem + 2 * ACT_HALVES;
  _Float16* wtb1 = wtb0 + WT_HALVES;
  const int p0 = blockIdx.x * 128;

  stage_async(panels + P_W0, wtb0, 16384);

  for (int i = threadIdx.x; i < 128 * 64; i += blockDim.x) {
    int m = i >> 6, c = i & 63;
    float v = (c < IN_CH) ? x[(size_t)(p0 + m) * (IN_CH + IN_VIEWS) + c] : 0.f;
    actA[m * 64 + swz(m, c)] = (_Float16)v;
  }

  wait_async_and_sync();
  stage_async(panels + P_WS(0, 0), wtb1, WT_HALVES);
  mlp_layer<2, 16, true>(actA, 64, actB, 256, wtb0, 64, b0, 0);

  _Float16* cur = actB; _Float16* nxt = actA;
  _Float16* wcur = wtb1; _Float16* wnxt = wtb0;
  for (int l = 0; l < 3; ++l) {
    for (int half = 0; half < 2; ++half) {
      wait_async_and_sync();
      int s = l * 2 + half;
      if (s < 5) stage_async(panels + P_WS((s + 1) >> 1, (s + 1) & 1), wnxt, WT_HALVES);
      mlp_layer<8, 8, true>(cur, 256, nxt, 256, wcur, 256, bs + l * WIDTH, half * 8);
      { _Float16* t = wcur; wcur = wnxt; wnxt = t; }
    }
    { _Float16* t = cur; cur = nxt; nxt = t; }
  }
  __syncthreads();
  for (int i = threadIdx.x; i < 128 * 256; i += blockDim.x) {
    int m = i >> 8, c = i & 255;
    h_out[(size_t)(p0 + m) * WIDTH + c] = cur[m * 256 + swz(m, c)];
  }
}

// ---------------- per-head branch ----------------
__global__ __launch_bounds__(256) void head_kernel(
    const float* __restrict__ x, const _Float16* __restrict__ h_in,
    const int* __restrict__ lists, const int* __restrict__ counts,
    const _Float16* __restrict__ panels,
    const float* __restrict__ tb0, const float* __restrict__ tbs,
    const float* __restrict__ featb, const float* __restrict__ alphab,
    const float* __restrict__ viewsb, const float* __restrict__ rgbb,
    float* __restrict__ out) {
  const int head = blockIdx.y;
  const int tile = blockIdx.x;
  const int cnt  = counts[head];
  if (tile * 128 >= cnt) return;

  extern __shared__ _Float16 smem[];
  _Float16* actA = smem;
  _Float16* actB = smem + ACT_HALVES;
  _Float16* wtb0 = smem + 2 * ACT_HALVES;
  _Float16* wtb1 = wtb0 + WT_HALVES;
  __shared__ int   pidx[128];
  __shared__ float alpha_s[128];

  const _Float16* PB = panels + P_HEADBASE + (size_t)head * PH_SIZE;
  _Float16* wcur = wtb0;
  _Float16* wnxt = wtb1;

  stage_async(PB + PH_TW0(0), wcur, 20480);

  if (threadIdx.x < 128) {
    int i = tile * 128 + threadIdx.x;
    pidx[threadIdx.x] = lists[(size_t)head * N_PTS + ((i < cnt) ? i : (cnt - 1))];
  }
  __syncthreads();

  // ht = [pts(63) | h(256) | 0] -> actA (stride 320), overlapped with first panel copy
  for (int i = threadIdx.x; i < 128 * 320; i += blockDim.x) {
    int m = i / 320, c = i - m * 320;
    int p = pidx[m];
    float v;
    if (c < IN_CH)              v = x[(size_t)p * (IN_CH + IN_VIEWS) + c];
    else if (c < IN_CH + WIDTH) v = (float)h_in[(size_t)p * WIDTH + (c - IN_CH)];
    else                        v = 0.f;
    actA[m * 320 + swz(m, c)] = (_Float16)v;
  }

  // tW0: 319 -> 256 in 4 N-quarters
  for (int q = 0; q < 4; ++q) {
    wait_async_and_sync();
    if (q < 3) stage_async(PB + PH_TW0(q + 1), wnxt, 20480);
    else       stage_async(PB + PH_TWS(0, 0), wnxt, WT_HALVES);
    mlp_layer<10, 4, true>(actA, 320, actB, 256, wcur, 320, tb0 + head * WIDTH, q * 4);
    { _Float16* t = wcur; wcur = wnxt; wnxt = t; }
  }

  _Float16* cur = actB; _Float16* nxt = actA;
  for (int l = 0; l < 3; ++l) {
    for (int half = 0; half < 2; ++half) {
      wait_async_and_sync();
      int s = l * 2 + half;
      if (s < 5) stage_async(PB + PH_TWS((s + 1) >> 1, (s + 1) & 1), wnxt, WT_HALVES);
      else       stage_async(PB + PH_ALPHA, wnxt, 4096);
      mlp_layer<8, 8, true>(cur, 256, nxt, 256, wcur, 256,
                            tbs + ((size_t)head * 3 + l) * WIDTH, half * 8);
      { _Float16* t = wcur; wcur = wnxt; wnxt = t; }
    }
    { _Float16* t = cur; cur = nxt; nxt = t; }
  }
  // cur == actA (ha), nxt == actB

  // alpha: 256 -> 1 (N padded to 16)
  wait_async_and_sync();
  stage_async(PB + PH_FEAT(0), wnxt, WT_HALVES);
  {
    const int lane = threadIdx.x & 31;
    const int wave = threadIdx.x >> 5;
    const int m0 = wave * 16;
    const int hi16 = lane >> 4;
    v16h a[8], bf[8];
#pragma unroll
    for (int kt = 0; kt < 8; ++kt)
      a[kt] = load_a_frag(cur, 256, m0 + (lane & 15), kt * 32, hi16 * 8);
#pragma unroll
    for (int kt = 0; kt < 8; ++kt)
      bf[kt] = load_b_frag(wcur, 256, lane & 15, kt * 32, hi16 * 16);
    v8f acc = {};
#pragma unroll
    for (int kt = 0; kt < 8; ++kt)
      acc = __builtin_amdgcn_wmma_f32_16x16x32_f16(false, a[kt], false, bf[kt], (short)0,
                                                   acc, false, false);
    if ((lane & 15) == 0) {
      float ab = alphab[head];
#pragma unroll
      for (int r = 0; r < 8; ++r) alpha_s[m0 + hi16 * 8 + r] = acc[r] + ab;
    }
  }
  { _Float16* t = wcur; wcur = wnxt; wnxt = t; }

  // feat: 256 -> 256, no relu, into actB cols 0..255 of stride 320
  for (int half = 0; half < 2; ++half) {
    wait_async_and_sync();
    if (half == 0) stage_async(PB + PH_FEAT(1), wnxt, WT_HALVES);
    else           stage_async(PB + PH_VIEWS(0), wnxt, 20480);
    mlp_layer<8, 8, false>(cur, 256, nxt, 320, wcur, 256, featb + head * WIDTH, half * 8);
    { _Float16* t = wcur; wcur = wnxt; wnxt = t; }
  }

  // append views (27 ch) at cols 256..282, zero to 319
  for (int i = threadIdx.x; i < 128 * 64; i += blockDim.x) {
    int m = i >> 6, j = i & 63, c = 256 + j;
    int p = pidx[m];
    float v = (j < IN_VIEWS) ? x[(size_t)p * (IN_CH + IN_VIEWS) + IN_CH + j] : 0.f;
    nxt[m * 320 + swz(m, c)] = (_Float16)v;
  }

  // views: 283 -> 128 in 2 N-halves, into actA stride 128
  for (int half = 0; half < 2; ++half) {
    wait_async_and_sync();
    if (half == 0) stage_async(PB + PH_VIEWS(1), wnxt, 20480);
    else           stage_async(PB + PH_RGB, wnxt, 2048);
    mlp_layer<9, 4, true>(nxt, 320, cur, 128, wcur, 320, viewsb + head * 128, half * 4);
    { _Float16* t = wcur; wcur = wnxt; wnxt = t; }
  }

  // rgb: 128 -> 3; scatter [r,g,b,alpha]
  wait_async_and_sync();
  {
    const int lane = threadIdx.x & 31;
    const int wave = threadIdx.x >> 5;
    const int m0 = wave * 16;
    const int hi16 = lane >> 4;
    const int nlane = lane & 15;
    v16h a[4], bf[4];
#pragma unroll
    for (int kt = 0; kt < 4; ++kt)
      a[kt] = load_a_frag(cur, 128, m0 + nlane, kt * 32, hi16 * 8);
#pragma unroll
    for (int kt = 0; kt < 4; ++kt)
      bf[kt] = load_b_frag(wcur, 128, nlane, kt * 32, hi16 * 16);
    v8f acc = {};
#pragma unroll
    for (int kt = 0; kt < 4; ++kt)
      acc = __builtin_amdgcn_wmma_f32_16x16x32_f16(false, a[kt], false, bf[kt], (short)0,
                                                   acc, false, false);
    float rb = (nlane < 3) ? rgbb[head * 3 + nlane] : 0.f;
#pragma unroll
    for (int r = 0; r < 8; ++r) {
      int m = m0 + hi16 * 8 + r;
      int gi = tile * 128 + m;
      if (gi < cnt) {
        int p = pidx[m];
        if (nlane < 3)       out[(size_t)p * 4 + nlane] = acc[r] + rb;
        else if (nlane == 3) out[(size_t)p * 4 + 3]     = alpha_s[m];
      }
    }
  }
}

extern "C" void kernel_launch(void* const* d_in, const int* in_sizes, int n_in,
                              void* d_out, int out_size, void* d_ws, size_t ws_size,
                              hipStream_t stream) {
  const float* x        = (const float*)d_in[0];
  const int*   head_idx = (const int*)d_in[1];
  const float* W0       = (const float*)d_in[2];
  const float* b0       = (const float*)d_in[3];
  const float* Ws       = (const float*)d_in[4];
  const float* bs       = (const float*)d_in[5];
  const float* tW0      = (const float*)d_in[6];
  const float* tb0      = (const float*)d_in[7];
  const float* tWs      = (const float*)d_in[8];
  const float* tbs      = (const float*)d_in[9];
  const float* featW    = (const float*)d_in[10];
  const float* featb    = (const float*)d_in[11];
  const float* alphaW   = (const float*)d_in[12];
  const float* alphab   = (const float*)d_in[13];
  const float* viewsW   = (const float*)d_in[14];
  const float* viewsb   = (const float*)d_in[15];
  const float* rgbW     = (const float*)d_in[16];
  const float* rgbb     = (const float*)d_in[17];
  float* out = (float*)d_out;
  (void)in_sizes; (void)n_in; (void)out_size; (void)ws_size;

  // ws: h (N x 256 f16) | panels (PANELS_TOTAL halves) | counts | lists
  char* base = (char*)d_ws;
  _Float16* h_ws   = (_Float16*)base;
  _Float16* panels = (_Float16*)(base + (size_t)N_PTS * WIDTH * 2);
  int* counts = (int*)(base + (size_t)N_PTS * WIDTH * 2 + (size_t)PANELS_TOTAL * 2);
  int* lists  = counts + 16;

  PanelTable tab{};
  int c = 0;
  auto add = [&](const float* src, unsigned off, int Kin, int Ntot, int n0,
                 int nthis, int Npad, int kpad) {
    tab.d[c].src = src; tab.d[c].dstoff = off;
    tab.d[c].Kin = (short)Kin; tab.d[c].Ntot = (short)Ntot;
    tab.d[c].n0 = (short)n0; tab.d[c].nthis = (short)nthis;
    tab.d[c].Npad = (short)Npad; tab.d[c].kpad = (short)kpad;
    ++c;
  };
  add(W0, P_W0, IN_CH, WIDTH, 0, 256, 256, 64);
  for (int l = 0; l < 3; ++l)
    for (int h2 = 0; h2 < 2; ++h2)
      add(Ws + (size_t)l * WIDTH * WIDTH, P_WS(l, h2), 256, 256, h2 * 128, 128, 128, 256);
  for (int hd = 0; hd < NHEADS; ++hd) {
    unsigned B = P_HEADBASE + (unsigned)hd * PH_SIZE;
    for (int q = 0; q < 4; ++q)
      add(tW0 + (size_t)hd * 319 * WIDTH, B + PH_TW0(q), 319, 256, q * 64, 64, 64, 320);
    for (int l = 0; l < 3; ++l)
      for (int h2 = 0; h2 < 2; ++h2)
        add(tWs + ((size_t)hd * 3 + l) * WIDTH * WIDTH, B + PH_TWS(l, h2),
            256, 256, h2 * 128, 128, 128, 256);
    add(alphaW + (size_t)hd * WIDTH, B + PH_ALPHA, 256, 1, 0, 1, 16, 256);
    for (int h2 = 0; h2 < 2; ++h2)
      add(featW + (size_t)hd * WIDTH * WIDTH, B + PH_FEAT(h2), 256, 256, h2 * 128, 128, 128, 256);
    for (int h2 = 0; h2 < 2; ++h2)
      add(viewsW + (size_t)hd * 283 * 128, B + PH_VIEWS(h2), 283, 128, h2 * 64, 64, 64, 320);
    add(rgbW + (size_t)hd * 128 * 3, B + PH_RGB, 128, 3, 0, 3, 16, 128);
  }
  tab.count = c;  // 71

  hipFuncSetAttribute((const void*)trunk_kernel,
                      hipFuncAttributeMaxDynamicSharedMemorySize, SMEM_BYTES);
  hipFuncSetAttribute((const void*)head_kernel,
                      hipFuncAttributeMaxDynamicSharedMemorySize, SMEM_BYTES);

  prep_panels_kernel<<<c, 256, 0, stream>>>(tab, panels);
  route_init_kernel<<<1, 32, 0, stream>>>(counts);
  route_scatter_kernel<<<N_PTS / 256, 256, 0, stream>>>(head_idx, counts, lists);
  trunk_kernel<<<N_PTS / 128, 256, SMEM_BYTES, stream>>>(x, b0, bs, panels, h_ws);
  head_kernel<<<dim3(N_PTS / 128, NHEADS), 256, SMEM_BYTES, stream>>>(
      x, h_ws, lists, counts, panels, tb0, tbs, featb, alphab, viewsb, rgbb, out);
}
